// DIRAC_76020921140148
// MI455X (gfx1250) — matrix operand order repaired and checked by reading, not compiled, not run on hardware
//
#include <hip/hip_runtime.h>
#include <hip/hip_bf16.h>

typedef __attribute__((ext_vector_type(16))) _Float16 v16h;
typedef __attribute__((ext_vector_type(8)))  float    v8f;

#define NN 100000
#define NE 3200000
#define BS 256
#define XS 8   // padded node-feature row stride (6 used + 2 zero), 32B aligned

__device__ __forceinline__ float reluf(float v)  { return v > 0.f ? v : 0.f; }
__device__ __forceinline__ float lreluf(float v) { return v > 0.f ? v : 0.2f * v; }
__device__ __forceinline__ float max3f(float a, float b, float c) { return fmaxf(a, fmaxf(b, c)); }

__device__ __forceinline__ v8f wmma_f16(v16h a, v16h b, v8f c) {
  // D = A(16x32 f16) x B(32x16 f16) + C(16x16 f32)
  return __builtin_amdgcn_wmma_f32_16x16x32_f16(false, a, false, b, (short)0, c, false, false);
}

// gather one padded node row (6 valid floats) with two aligned b128 loads
__device__ __forceinline__ void load_row6(const float* __restrict__ t, int n, float r[6]) {
  const float4* t4 = (const float4*)t;
  float4 lo = t4[n * 2];
  float4 hi = t4[n * 2 + 1];
  r[0] = lo.x; r[1] = lo.y; r[2] = lo.z; r[3] = lo.w; r[4] = hi.x; r[5] = hi.y;
}

// ---------------------------------------------------------------- utilities
__global__ void zero_kernel(float* __restrict__ p, int n) {
  int i = blockIdx.x * BS + threadIdx.x;
  if (i < n) p[i] = 0.f;
}

// ------------------------------------------------------- edge layer 1 (fused)
// m = x[s]+x[d] (5), hx = wx(10x5) m + bx, he = we(2x1)*ea + be,
// relu+maxpool3 over concat(12) -> 4; store ea1 (float4), atomic seg-sum agg1.
__global__ __launch_bounds__(BS)
void edge1_kernel(const float* __restrict__ x, const int* __restrict__ ei,
                  const float* __restrict__ ea,
                  const float* __restrict__ wx, const float* __restrict__ bx,
                  const float* __restrict__ we, const float* __restrict__ be,
                  float4* __restrict__ ea1, float* __restrict__ agg1) {
  int e = blockIdx.x * BS + threadIdx.x;
  if (e >= NE) return;
  int s = ei[e], d = ei[NE + e];
  float m[5];
#pragma unroll
  for (int c = 0; c < 5; ++c) m[c] = x[s * 5 + c] + x[d * 5 + c];
  float hx[10];
#pragma unroll
  for (int o = 0; o < 10; ++o) {
    float acc = bx[o];
#pragma unroll
    for (int c = 0; c < 5; ++c) acc += wx[o * 5 + c] * m[c];
    hx[o] = acc;
  }
  float a0  = ea[e];
  float he0 = we[0] * a0 + be[0];
  float he1 = we[1] * a0 + be[1];
  float o0 = reluf(max3f(hx[0], hx[1], hx[2]));
  float o1 = reluf(max3f(hx[3], hx[4], hx[5]));
  float o2 = reluf(max3f(hx[6], hx[7], hx[8]));
  float o3 = reluf(max3f(hx[9], he0, he1));
  ea1[e] = make_float4(o0, o1, o2, o3);
  atomicAdd(&agg1[s * 4 + 0], o0);
  atomicAdd(&agg1[s * 4 + 1], o1);
  atomicAdd(&agg1[s * 4 + 2], o2);
  atomicAdd(&agg1[s * 4 + 3], o3);
}

// ------------------------------------------------------------- node layer 1
__global__ __launch_bounds__(BS)
void node1_kernel(const float* __restrict__ x, const float* __restrict__ agg1,
                  const float* __restrict__ wx, const float* __restrict__ bx,
                  const float* __restrict__ we, const float* __restrict__ be,
                  float* __restrict__ xout) {
  int n = blockIdx.x * BS + threadIdx.x;
  if (n >= NN) return;
  float v[XS] = {0.f, 0.f, 0.f, 0.f, 0.f, 0.f, 0.f, 0.f};
#pragma unroll
  for (int o = 0; o < 3; ++o) {
    float a = bx[o];
#pragma unroll
    for (int c = 0; c < 5; ++c) a += wx[o * 5 + c] * x[n * 5 + c];
    float b = be[o];
#pragma unroll
    for (int c = 0; c < 4; ++c) b += we[o * 4 + c] * agg1[n * 4 + c];
    v[o]     = reluf(a);
    v[3 + o] = reluf(b);
  }
  float4* xo4 = (float4*)xout;
  xo4[n * 2]     = make_float4(v[0], v[1], v[2], v[3]);
  xo4[n * 2 + 1] = make_float4(v[4], v[5], 0.f, 0.f);
}

// -------------------------------------------------- edge layers 2/3 (fused)
// m = xin[s]+xin[d] (6, padded rows), hx = wx(18x6) m, he = we(12xCIN) eain,
// relu+maxpool3 over concat(30) -> 10; optional ea store; atomic segment-sum.
template <int CIN, bool WRITE_EA>
__global__ __launch_bounds__(BS)
void edge_big_kernel(const float* __restrict__ xin, const int* __restrict__ ei,
                     const float* __restrict__ eain,
                     const float* __restrict__ wx, const float* __restrict__ bx,
                     const float* __restrict__ we, const float* __restrict__ be,
                     float* __restrict__ eaout, float* __restrict__ agg) {
  __shared__ float swx[108], sbx[18], swe[12 * CIN], sbe[12];
  for (int i = threadIdx.x; i < 108; i += BS) swx[i] = wx[i];
  for (int i = threadIdx.x; i < 18; i += BS) sbx[i] = bx[i];
  for (int i = threadIdx.x; i < 12 * CIN; i += BS) swe[i] = we[i];
  for (int i = threadIdx.x; i < 12; i += BS) sbe[i] = be[i];
  __syncthreads();

  int e = blockIdx.x * BS + threadIdx.x;
  if (e >= NE) return;
  int s = ei[e], d = ei[NE + e];
  float rs[6], rd[6], m[6];
  load_row6(xin, s, rs);
  load_row6(xin, d, rd);
#pragma unroll
  for (int c = 0; c < 6; ++c) m[c] = rs[c] + rd[c];

  float ein[CIN];
  if constexpr (CIN == 4) {
    float4 t = ((const float4*)eain)[e];
    ein[0] = t.x; ein[1] = t.y; ein[2] = t.z; ein[3] = t.w;
  } else {
#pragma unroll
    for (int c = 0; c < CIN; ++c) ein[c] = eain[(size_t)e * CIN + c];
  }

  float out[10];
#pragma unroll
  for (int g = 0; g < 6; ++g) {  // pooled groups from hx (18 -> 6)
    float mx = -3.402823466e+38f;
#pragma unroll
    for (int j = 0; j < 3; ++j) {
      int o = g * 3 + j;
      float acc = sbx[o];
#pragma unroll
      for (int c = 0; c < 6; ++c) acc += swx[o * 6 + c] * m[c];
      mx = fmaxf(mx, acc);
    }
    out[g] = reluf(mx);
  }
#pragma unroll
  for (int g = 0; g < 4; ++g) {  // pooled groups from he (12 -> 4)
    float mx = -3.402823466e+38f;
#pragma unroll
    for (int j = 0; j < 3; ++j) {
      int o = g * 3 + j;
      float acc = sbe[o];
#pragma unroll
      for (int c = 0; c < CIN; ++c) acc += swe[o * CIN + c] * ein[c];
      mx = fmaxf(mx, acc);
    }
    out[6 + g] = reluf(mx);
  }
  if (WRITE_EA) {
#pragma unroll
    for (int g = 0; g < 10; ++g) eaout[(size_t)e * 10 + g] = out[g];
  }
#pragma unroll
  for (int g = 0; g < 10; ++g) atomicAdd(&agg[s * 10 + g], out[g]);
}

// ------------------------------------------------------- node layers 2/3
// LEAKY variant also block-reduces the x3 node-sum into global state[6].
template <bool LEAKY>
__global__ __launch_bounds__(BS)
void node_big_kernel(const float* __restrict__ xin, const float* __restrict__ agg,
                     const float* __restrict__ wx, const float* __restrict__ bx,
                     const float* __restrict__ we, const float* __restrict__ be,
                     float* __restrict__ xout, float* __restrict__ state) {
  int n = blockIdx.x * BS + threadIdx.x;
  float v[6] = {0.f, 0.f, 0.f, 0.f, 0.f, 0.f};
  if (n < NN) {
    float r[6];
    load_row6(xin, n, r);
#pragma unroll
    for (int o = 0; o < 3; ++o) {
      float a = bx[o];
#pragma unroll
      for (int c = 0; c < 6; ++c) a += wx[o * 6 + c] * r[c];
      float b = be[o];
#pragma unroll
      for (int c = 0; c < 10; ++c) b += we[o * 10 + c] * agg[n * 10 + c];
      v[o]     = LEAKY ? lreluf(a) : reluf(a);
      v[3 + o] = LEAKY ? lreluf(b) : reluf(b);
    }
    float4* xo4 = (float4*)xout;
    xo4[n * 2]     = make_float4(v[0], v[1], v[2], v[3]);
    xo4[n * 2 + 1] = make_float4(v[4], v[5], 0.f, 0.f);
  }
  if constexpr (LEAKY) {
    __shared__ float sred[6];
    if (threadIdx.x < 6) sred[threadIdx.x] = 0.f;
    __syncthreads();
    if (n < NN) {
#pragma unroll
      for (int k = 0; k < 6; ++k) atomicAdd(&sred[k], v[k]);
    }
    __syncthreads();
    if (threadIdx.x < 6) atomicAdd(&state[threadIdx.x], sred[threadIdx.x]);
  }
}

// -------------------------------------------------------- Q-head via WMMA
// One wave (32 threads) per 16-node tile. fc1: 3x wmma (K=12 padded to 32),
// LDS relayout, fc2: 2x wmma (K=36 = 32+4), fc3: shuffle-reduced dot.
__global__ __launch_bounds__(32)
void qhead_kernel(const float* __restrict__ x3, const float* __restrict__ state,
                  const float* __restrict__ w1, const float* __restrict__ b1,
                  const float* __restrict__ w2, const float* __restrict__ b2,
                  const float* __restrict__ w3, const float* __restrict__ b3,
                  float* __restrict__ out) {
  __shared__ float h1s[16][40];  // 16 rows x 36 (padded)
  const int lane = threadIdx.x;   // 0..31
  const int half = lane >> 4;     // 0 or 1
  const int lm   = lane & 15;
  const int tile = blockIdx.x;
  const int node = tile * 16 + lm;  // row M = lm

  // q12 = [state(6), x3(6)] for row M = lm
  float q12[12];
#pragma unroll
  for (int k = 0; k < 6; ++k) q12[k] = state[k];
  {
    const float4* x4 = (const float4*)x3;
    float4 lo = x4[node * 2];
    float4 hi = x4[node * 2 + 1];
    q12[6] = lo.x; q12[7] = lo.y; q12[8] = lo.z;
    q12[9] = lo.w; q12[10] = hi.x; q12[11] = hi.y;
  }

  // A fragment (16x32 f16): lanes 0-15 hold K0..15 (elems 0-15); lanes 16-31
  // hold K16..31 (zero pad K>=12).
  v16h a1 = {};
  if (half == 0) {
#pragma unroll
    for (int j = 0; j < 12; ++j) a1[j] = (_Float16)q12[j];
  }

  // fc1: out cols 0..35 in three 16-col B tiles. B[k][n] = w1[n*12+k].
  v8f acc0 = {}, acc1 = {}, acc2 = {};
  {
    v16h b = {};
    int col = lm;
    if (half == 0) {
#pragma unroll
      for (int j = 0; j < 12; ++j) b[j] = (_Float16)w1[col * 12 + j];
    }
    acc0 = wmma_f16(a1, b, acc0);
  }
  {
    v16h b = {};
    int col = 16 + lm;
    if (half == 0) {
#pragma unroll
      for (int j = 0; j < 12; ++j) b[j] = (_Float16)w1[col * 12 + j];
    }
    acc1 = wmma_f16(a1, b, acc1);
  }
  {
    v16h b = {};
    int col = 32 + lm;
    if (half == 0 && col < 36) {
#pragma unroll
      for (int j = 0; j < 12; ++j) b[j] = (_Float16)w1[col * 12 + j];
    }
    acc2 = wmma_f16(a1, b, acc2);
  }

  // bias + relu, spill 16x36 tile to LDS (C layout: vgpr m -> row m+8*half)
  float bia0 = b1[lm];
  float bia1 = b1[16 + lm];
  float bia2 = (32 + lm < 36) ? b1[32 + lm] : 0.f;
#pragma unroll
  for (int m = 0; m < 8; ++m) {
    int M = m + 8 * half;
    h1s[M][lm]      = fmaxf(acc0[m] + bia0, 0.f);
    h1s[M][16 + lm] = fmaxf(acc1[m] + bia1, 0.f);
    if (32 + lm < 36) h1s[M][32 + lm] = fmaxf(acc2[m] + bia2, 0.f);
  }
  __syncthreads();

  // fc2: A2 = 16x36 (two K-chunks: 0..31 and 32..63 with only 32..35 valid)
  v16h a20 = {}, a21 = {};
  {
    int M = lm;
    if (half == 0) {
#pragma unroll
      for (int j = 0; j < 16; ++j) a20[j] = (_Float16)h1s[M][j];        // K = 0..15
#pragma unroll
      for (int j = 0; j < 4; ++j) a21[j] = (_Float16)h1s[M][32 + j];  // K=32..35
    } else {
#pragma unroll
      for (int j = 0; j < 16; ++j) a20[j] = (_Float16)h1s[M][16 + j];   // K = 16..31
    }
  }
  v16h b20 = {}, b21 = {};
  {
    int nco = lm;
    bool cv = nco < 10;
    if (cv) {
      if (half == 0) {
#pragma unroll
        for (int j = 0; j < 16; ++j) b20[j] = (_Float16)w2[nco * 36 + j];       // K=0..15
#pragma unroll
        for (int j = 0; j < 4; ++j)  b21[j] = (_Float16)w2[nco * 36 + 32 + j];  // K=32..35
      } else {
#pragma unroll
        for (int j = 0; j < 16; ++j) b20[j] = (_Float16)w2[nco * 36 + 16 + j];  // K=16..31
      }
    }
  }
  v8f acc = {};
  acc = wmma_f16(a20, b20, acc);
  acc = wmma_f16(a21, b21, acc);

  // fc3: per-row dot over 10 cols; reduce across the 16 lanes of each half.
  bool cv = lm < 10;
  float bia = cv ? b2[lm] : 0.f;
  float w3n = cv ? w3[lm] : 0.f;
  float p[8];
#pragma unroll
  for (int m = 0; m < 8; ++m) p[m] = fmaxf(acc[m] + bia, 0.f) * w3n;
#pragma unroll
  for (int mask = 1; mask < 16; mask <<= 1) {
#pragma unroll
    for (int m = 0; m < 8; ++m) p[m] += __shfl_xor(p[m], mask, 32);
  }
  float b3v = b3[0];
  if (lm < 8) {
    float wv = 0.f;
#pragma unroll
    for (int m = 0; m < 8; ++m)
      if (lm == m) wv = p[m];
    out[tile * 16 + lm + 8 * half] = lreluf(wv + b3v);
  }
}

// ------------------------------------------------------------------- launch
extern "C" void kernel_launch(void* const* d_in, const int* in_sizes, int n_in,
                              void* d_out, int out_size, void* d_ws, size_t ws_size,
                              hipStream_t stream) {
  (void)n_in; (void)out_size; (void)ws_size;
  const int N = NN, E = NE;

  int iX, iEI, iEA,
      iE1WX, iE1BX, iE1WE, iE1BE, iN1WX, iN1BX, iN1WE, iN1BE,
      iE2WX, iE2BX, iE2WE, iE2BE, iN2WX, iN2BX, iN2WE, iN2BE,
      iE3WX, iE3BX, iE3WE, iE3BE, iN3WX, iN3BX, iN3WE, iN3BE,
      iF1W, iF1B, iF2W, iF2B, iF3W, iF3B;

  if (in_sizes[0] == N * 5) {
    // dict-insertion order: x, edge_index, edge_attr, params{edge1{wx,bx,we,be},...}
    iX = 0; iEI = 1; iEA = 2;
    iE1WX = 3;  iE1BX = 4;  iE1WE = 5;  iE1BE = 6;
    iN1WX = 7;  iN1BX = 8;  iN1WE = 9;  iN1BE = 10;
    iE2WX = 11; iE2BX = 12; iE2WE = 13; iE2BE = 14;
    iN2WX = 15; iN2BX = 16; iN2WE = 17; iN2BE = 18;
    iE3WX = 19; iE3BX = 20; iE3WE = 21; iE3BE = 22;
    iN3WX = 23; iN3BX = 24; iN3WE = 25; iN3BE = 26;
    iF1W = 27; iF1B = 28; iF2W = 29; iF2B = 30; iF3W = 31; iF3B = 32;
  } else {
    // jax-pytree sorted-key order: edge_attr, edge_index, params(sorted), x
    iEA = 0; iEI = 1;
    iE1BE = 2;  iE1BX = 3;  iE1WE = 4;  iE1WX = 5;
    iE2BE = 6;  iE2BX = 7;  iE2WE = 8;  iE2WX = 9;
    iE3BE = 10; iE3BX = 11; iE3WE = 12; iE3WX = 13;
    iF1B = 14; iF1W = 15; iF2B = 16; iF2W = 17; iF3B = 18; iF3W = 19;
    iN1BE = 20; iN1BX = 21; iN1WE = 22; iN1WX = 23;
    iN2BE = 24; iN2BX = 25; iN2WE = 26; iN2WX = 27;
    iN3BE = 28; iN3BX = 29; iN3WE = 30; iN3WX = 31;
    iX = 32;
  }

  const float* x   = (const float*)d_in[iX];
  const int*   ei  = (const int*)d_in[iEI];
  const float* ea  = (const float*)d_in[iEA];
  float* out = (float*)d_out;

  // workspace carve-up (64-float aligned regions)
  float* ws = (float*)d_ws;
  size_t off = 0;
  auto carve = [&](size_t nf) { float* p = ws + off; off += (nf + 63) & ~size_t(63); return p; };
  float* agg1  = carve((size_t)N * 4);
  float* agg2  = carve((size_t)N * 10);
  float* agg3  = carve((size_t)N * 10);
  float* state = carve(16);
  size_t zeroN = off;  // contiguous zero region: agg1..state
  float* x1 = carve((size_t)N * XS);
  float* x2 = carve((size_t)N * XS);
  float* x3 = carve((size_t)N * XS);
  float* ea1 = carve((size_t)E * 4);
  float* ea2 = carve((size_t)E * 10);

  const unsigned gz = (unsigned)((zeroN + BS - 1) / BS);
  const unsigned ge = (unsigned)((E + BS - 1) / BS);
  const unsigned gn = (unsigned)((N + BS - 1) / BS);
  const unsigned gq = (unsigned)(N / 16);  // 6250 tiles exactly

  zero_kernel<<<gz, BS, 0, stream>>>(ws, (int)zeroN);

  edge1_kernel<<<ge, BS, 0, stream>>>(
      x, ei, ea,
      (const float*)d_in[iE1WX], (const float*)d_in[iE1BX],
      (const float*)d_in[iE1WE], (const float*)d_in[iE1BE],
      (float4*)ea1, agg1);

  node1_kernel<<<gn, BS, 0, stream>>>(
      x, agg1,
      (const float*)d_in[iN1WX], (const float*)d_in[iN1BX],
      (const float*)d_in[iN1WE], (const float*)d_in[iN1BE], x1);

  edge_big_kernel<4, true><<<ge, BS, 0, stream>>>(
      x1, ei, ea1,
      (const float*)d_in[iE2WX], (const float*)d_in[iE2BX],
      (const float*)d_in[iE2WE], (const float*)d_in[iE2BE], ea2, agg2);

  node_big_kernel<false><<<gn, BS, 0, stream>>>(
      x1, agg2,
      (const float*)d_in[iN2WX], (const float*)d_in[iN2BX],
      (const float*)d_in[iN2WE], (const float*)d_in[iN2BE], x2, nullptr);

  edge_big_kernel<10, false><<<ge, BS, 0, stream>>>(
      x2, ei, ea2,
      (const float*)d_in[iE3WX], (const float*)d_in[iE3BX],
      (const float*)d_in[iE3WE], (const float*)d_in[iE3BE], nullptr, agg3);

  node_big_kernel<true><<<gn, BS, 0, stream>>>(
      x2, agg3,
      (const float*)d_in[iN3WX], (const float*)d_in[iN3BX],
      (const float*)d_in[iN3WE], (const float*)d_in[iN3BE], x3, state);

  qhead_kernel<<<gq, 32, 0, stream>>>(
      x3, state,
      (const float*)d_in[iF1W], (const float*)d_in[iF1B],
      (const float*)d_in[iF2W], (const float*)d_in[iF2B],
      (const float*)d_in[iF3W], (const float*)d_in[iF3B], out);
}
